// VQ_VAE_8538394984802
// MI455X (gfx1250) — matrix operand order repaired and checked by reading, not compiled, not run on hardware
//
#include <hip/hip_runtime.h>
#include <hip/hip_bf16.h>
#include <math.h>
#include <stdint.h>

typedef __attribute__((ext_vector_type(16))) __bf16 v16bf;
typedef __attribute__((ext_vector_type(8)))  __bf16 v8bf;
typedef __attribute__((ext_vector_type(8)))  float  v8f;
typedef __attribute__((ext_vector_type(4)))  int    v4i;

union FragAB { v16bf v; v8bf h[2]; };

#define BM 128
#define BN 64
#define BK 32
#define LDP (BK + 8)   // LDS row stride in halfs (80B: 16B-aligned, staggers banks)

#if __has_builtin(__builtin_amdgcn_global_load_async_to_lds_b128)
#define HAS_ASYNC_LDS 1
#else
#define HAS_ASYNC_LDS 0
#endif

// copy 16B global -> LDS (async on gfx1250; direct staging fallback)
__device__ __forceinline__ void cp_g2l_b128(const __bf16* g, __bf16* l) {
#if HAS_ASYNC_LDS
  __builtin_amdgcn_global_load_async_to_lds_b128((v4i*)g, (v4i*)l, 0, 0);
#else
  *(v8bf*)l = *(const v8bf*)g;
#endif
}

template <int N>
__device__ __forceinline__ void wait_async_lds() {
#if HAS_ASYNC_LDS
#if __has_builtin(__builtin_amdgcn_s_wait_asynccnt)
  __builtin_amdgcn_s_wait_asynccnt((short)N);
#else
  asm volatile("s_wait_asynccnt %0" ::"i"(N) : "memory");
#endif
#endif
}

// ------------------------------------------------- fp32 -> bf16 with zero pad
// out is (Rp x Cp) bf16; in is (R x C) fp32; pad region = 0.
__global__ __launch_bounds__(256)
void pad_convert_kernel(const float* __restrict__ in, __bf16* __restrict__ out,
                        int R, int C, int Rp, int Cp) {
  const long long i = (long long)blockIdx.x * blockDim.x + threadIdx.x;
  const long long total = (long long)Rp * Cp;
  if (i >= total) return;
  const int r = (int)(i / Cp);
  const int c = (int)(i % Cp);
  out[i] = (r < R && c < C) ? (__bf16)in[(size_t)r * C + c] : (__bf16)0.0f;
}

// ------------------------------------------------- bf16 WMMA GEMM, branch-free
// C = act(A[M,Kp] * Bw[rows,Kp]^T + bias[N]); OP: 0=none 1=relu 2=tanh.
// Kp multiple of 32, weight rows multiple of 64 (zero padded), M multiple of 128.
// OUT_BF16: write bf16 to Cb (width Npad, zeros for N<=col<Npad), else fp32 to Cf.
template <int OP, bool OUT_BF16>
__global__ __launch_bounds__(256)
void gemm_bf16(const __bf16* __restrict__ A, const __bf16* __restrict__ Bw,
               const float* __restrict__ bias, float* __restrict__ Cf,
               __bf16* __restrict__ Cb, int Kp, int N, int Npad) {
  __shared__ __attribute__((aligned(16))) __bf16 As[2][BM][LDP];
  __shared__ __attribute__((aligned(16))) __bf16 Bs[2][BN][LDP];

  const int tid  = threadIdx.x;
  const int lane = tid & 31;
  const int wid  = tid >> 5;
  const int wm   = (wid & 3) * 32;
  const int wn   = (wid >> 2) * 32;
  const int m0   = blockIdx.x * BM;
  const int n0   = blockIdx.y * BN;
  const int hsel = lane >> 4;
  const int l15  = lane & 15;

  // per-thread 16B copy slots: A tile = 128 rows x 4 chunks (2 per thread),
  // B tile = 64 rows x 4 chunks (1 per thread)
  const int rowA = tid >> 2;       // 0..63
  const int c16  = tid & 3;        // chunk-of-16B in row
  const __bf16* Abase = A + (size_t)(m0 + rowA) * Kp + c16 * 8;
  const __bf16* Bbase = Bw + (size_t)(n0 + rowA) * Kp + c16 * 8;
  const size_t Aoff64 = (size_t)64 * Kp;

  v8f zero = {};
  v8f acc[2][2] = {{zero, zero}, {zero, zero}};

  const int nk = Kp >> 5;

  // prologue: stage chunk 0 into buffer 0
  cp_g2l_b128(Abase, &As[0][rowA][c16 * 8]);
  cp_g2l_b128(Abase + Aoff64, &As[0][rowA + 64][c16 * 8]);
  cp_g2l_b128(Bbase, &Bs[0][rowA][c16 * 8]);

  int buf = 0;
  for (int i = 0; i < nk; ++i) {
    if (i + 1 < nk) {
      const int kn = (i + 1) << 5;
      cp_g2l_b128(Abase + kn, &As[buf ^ 1][rowA][c16 * 8]);
      cp_g2l_b128(Abase + Aoff64 + kn, &As[buf ^ 1][rowA + 64][c16 * 8]);
      cp_g2l_b128(Bbase + kn, &Bs[buf ^ 1][rowA][c16 * 8]);
      wait_async_lds<3>();   // in-order: <=3 outstanding => current buffer landed
    } else {
      wait_async_lds<0>();
    }
    __syncthreads();

    // fragments per ISA 7.12.2 layouts; 2x2 WMMA with operand reuse
    FragAB a[2], b[2];
#pragma unroll
    for (int ii = 0; ii < 2; ++ii) {
      const int r = wm + ii * 16 + l15;  // A: K split {0..7,16..23} / {8..15,24..31}
      a[ii].h[0] = *(const v8bf*)&As[buf][r][hsel * 8];
      a[ii].h[1] = *(const v8bf*)&As[buf][r][16 + hsel * 8];
    }
#pragma unroll
    for (int jj = 0; jj < 2; ++jj) {
      const int r = wn + jj * 16 + l15;  // B: K split {0..7,16..23} / {8..15,24..31}
      b[jj].h[0] = *(const v8bf*)&Bs[buf][r][hsel * 8];
      b[jj].h[1] = *(const v8bf*)&Bs[buf][r][16 + hsel * 8];
    }
#pragma unroll
    for (int ii = 0; ii < 2; ++ii)
#pragma unroll
      for (int jj = 0; jj < 2; ++jj)
        acc[ii][jj] = __builtin_amdgcn_wmma_f32_16x16x32_bf16(
            false, a[ii].v, false, b[jj].v, (short)0, acc[ii][jj], false, false);
    __syncthreads();
    buf ^= 1;
  }

  // epilogue: C/D layout = VGPR r holds rows r (lanes 0-15) / r+8 (lanes 16-31)
#pragma unroll
  for (int ii = 0; ii < 2; ++ii) {
#pragma unroll
    for (int jj = 0; jj < 2; ++jj) {
      const int col  = n0 + wn + jj * 16 + l15;
      const int rowb = m0 + wm + ii * 16 + hsel * 8;
      if (OUT_BF16) {
        if (col < Npad) {
          const float bv = (col < N) ? bias[col] : 0.0f;
#pragma unroll
          for (int r = 0; r < 8; ++r) {
            float v = (col < N) ? acc[ii][jj][r] + bv : 0.0f;
            if (OP == 1) v = v > 0.0f ? v : 0.0f;
            if (OP == 2) v = tanhf(v);
            Cb[(size_t)(rowb + r) * Npad + col] = (__bf16)v;
          }
        }
      } else {
        if (col < N) {
          const float bv = bias[col];
#pragma unroll
          for (int r = 0; r < 8; ++r) {
            float v = acc[ii][jj][r] + bv;
            if (OP == 1) v = v > 0.0f ? v : 0.0f;
            if (OP == 2) v = tanhf(v);
            Cf[(size_t)(rowb + r) * N + col] = v;
          }
        }
      }
    }
  }
}

// ------------------------------------------------- VQ nearest-codeword
// z_e flat (B,200) viewed as (B,K=10,G=20): sub-vector g = cols k*20+g.
// Writes emb fp32 (d_out) and bf16 padded copy (width 224) for the decoder.
__global__ __launch_bounds__(256)
void quantize_kernel(const float* __restrict__ z_e, const float* __restrict__ codebook,
                     float* __restrict__ emb, __bf16* __restrict__ embb, int total) {
  __shared__ float cb[100];
  if (threadIdx.x < 100) cb[threadIdx.x] = codebook[threadIdx.x];
  __syncthreads();
  const int t = blockIdx.x * blockDim.x + threadIdx.x;
  if (t >= total) return;
  const int b = t / 20;
  const int g = t % 20;
  const size_t base  = (size_t)b * 200 + g;
  const size_t baseb = (size_t)b * 224;
  float v[10];
#pragma unroll
  for (int k = 0; k < 10; ++k) v[k] = z_e[base + k * 20];
  float best = 3.4e38f;
  int bi = 0;
#pragma unroll
  for (int c = 0; c < 10; ++c) {
    float d = 0.0f;
#pragma unroll
    for (int k = 0; k < 10; ++k) {
      const float df = v[k] - cb[c * 10 + k];
      d += df * df;
    }
    if (d < best) { best = d; bi = c; }
  }
#pragma unroll
  for (int k = 0; k < 10; ++k) {
    const float q = cb[bi * 10 + k];
    emb[base + k * 20] = q;
    embb[baseb + g + k * 20] = (__bf16)q;
  }
  // zero the 24 pad columns (200..223) of this row
  embb[baseb + 200 + g] = (__bf16)0.0f;
  if (g < 4) embb[baseb + 220 + g] = (__bf16)0.0f;
}

// ------------------------------------------------- launch
extern "C" void kernel_launch(void* const* d_in, const int* in_sizes, int n_in,
                              void* d_out, int out_size, void* d_ws, size_t ws_size,
                              hipStream_t stream) {
  const float* x   = (const float*)d_in[0];
  const float* W1  = (const float*)d_in[1];
  const float* b1  = (const float*)d_in[2];
  const float* W2  = (const float*)d_in[3];
  const float* b2  = (const float*)d_in[4];
  const float* W3  = (const float*)d_in[5];
  const float* b3  = (const float*)d_in[6];
  const float* W4  = (const float*)d_in[7];
  const float* b4  = (const float*)d_in[8];
  const float* cbk = (const float*)d_in[9];

  const int B = 65536;
  float* out   = (float*)d_out;
  float* recon = out;                    // (B,784)
  float* z_e   = out + (size_t)B * 784;  // (B,200)
  float* emb   = z_e + (size_t)B * 200;  // (B,200) == forward z_q

  // workspace (bf16, all K padded to 32, weight rows padded to 64):
  __bf16* base = (__bf16*)d_ws;
  size_t o = 0;
  __bf16* W1b = base + o; o += (size_t)448 * 800;   // W1: 400x784 -> 448x800
  __bf16* W2b = base + o; o += (size_t)256 * 416;   // W2: 200x400 -> 256x416
  __bf16* W3b = base + o; o += (size_t)448 * 224;   // W3: 400x200 -> 448x224
  __bf16* W4b = base + o; o += (size_t)832 * 416;   // W4: 784x400 -> 832x416
  __bf16* xb  = base + o; o += (size_t)B * 800;     // x:  Bx784  -> Bx800
  __bf16* hb  = base + o; o += (size_t)B * 416;     // h1 / h3 (reused), Bx416
  __bf16* qb  = base + o; o += (size_t)B * 224;     // emb bf16, Bx224

  auto padconv = [&](const float* in, __bf16* outp, int R, int C, int Rp, int Cp) {
    const long long n = (long long)Rp * Cp;
    pad_convert_kernel<<<(unsigned)((n + 255) / 256), 256, 0, stream>>>(in, outp, R, C, Rp, Cp);
  };
  padconv(W1, W1b, 400, 784, 448, 800);
  padconv(W2, W2b, 200, 400, 256, 416);
  padconv(W3, W3b, 400, 200, 448, 224);
  padconv(W4, W4b, 784, 400, 832, 416);
  padconv(x,  xb,  B, 784, B, 800);

  const dim3 blk(256);
  // h1 = relu(x @ W1^T + b1) -> bf16 (B,416)
  gemm_bf16<1, true><<<dim3(B / BM, 448 / BN), blk, 0, stream>>>(
      xb, W1b, b1, nullptr, hb, 800, 400, 416);
  // z_e = h1 @ W2^T + b2 -> fp32 (B,200) straight to d_out
  gemm_bf16<0, false><<<dim3(B / BM, 256 / BN), blk, 0, stream>>>(
      hb, W2b, b2, z_e, nullptr, 416, 200, 200);
  // emb = nearest-codeword(z_e): fp32 to d_out + bf16 padded copy
  quantize_kernel<<<(B * 20 + 255) / 256, 256, 0, stream>>>(z_e, cbk, emb, qb, B * 20);
  // h3 = relu(emb @ W3^T + b3) -> bf16 (B,416), reuse hb
  gemm_bf16<1, true><<<dim3(B / BM, 448 / BN), blk, 0, stream>>>(
      qb, W3b, b3, nullptr, hb, 224, 400, 416);
  // recon = tanh(h3 @ W4^T + b4) -> fp32 (B,784)
  gemm_bf16<2, false><<<dim3(B / BM, 832 / BN), blk, 0, stream>>>(
      hb, W4b, b4, recon, nullptr, 416, 784, 784);
}
